// PointNetPartSeg_66357244723316
// MI455X (gfx1250) — matrix operand order repaired and checked by reading, not compiled, not run on hardware
//
#include <hip/hip_runtime.h>

#define BB 16
#define NPTS 4096

typedef __attribute__((ext_vector_type(16))) _Float16 v16h;
typedef __attribute__((ext_vector_type(8)))  float    v8f;

// ---------------------------------------------------------------------------
// Fragment loaders: branch-free, 128-bit global loads, f32 -> f16 convert.
// Caller guarantees 16B alignment (lda % 32 floats == 0, base 256B aligned).
// ---------------------------------------------------------------------------
__device__ __forceinline__ v16h load_a_frag(const float* __restrict__ p1,
                                            const float* __restrict__ p2)
{
  float4 u0 = ((const float4*)p1)[0];
  float4 u1 = ((const float4*)p1)[1];
  float4 w0 = ((const float4*)p2)[0];
  float4 w1 = ((const float4*)p2)[1];
  v16h a;
  a[0]  = (_Float16)u0.x; a[1]  = (_Float16)u0.y; a[2]  = (_Float16)u0.z; a[3]  = (_Float16)u0.w;
  a[4]  = (_Float16)u1.x; a[5]  = (_Float16)u1.y; a[6]  = (_Float16)u1.z; a[7]  = (_Float16)u1.w;
  a[8]  = (_Float16)w0.x; a[9]  = (_Float16)w0.y; a[10] = (_Float16)w0.z; a[11] = (_Float16)w0.w;
  a[12] = (_Float16)w1.x; a[13] = (_Float16)w1.y; a[14] = (_Float16)w1.z; a[15] = (_Float16)w1.w;
  return a;
}

__device__ __forceinline__ v16h load_b_frag(const float* __restrict__ p)
{
  float4 u0 = ((const float4*)p)[0];
  float4 u1 = ((const float4*)p)[1];
  float4 u2 = ((const float4*)p)[2];
  float4 u3 = ((const float4*)p)[3];
  v16h b;
  b[0]  = (_Float16)u0.x; b[1]  = (_Float16)u0.y; b[2]  = (_Float16)u0.z; b[3]  = (_Float16)u0.w;
  b[4]  = (_Float16)u1.x; b[5]  = (_Float16)u1.y; b[6]  = (_Float16)u1.z; b[7]  = (_Float16)u1.w;
  b[8]  = (_Float16)u2.x; b[9]  = (_Float16)u2.y; b[10] = (_Float16)u2.z; b[11] = (_Float16)u2.w;
  b[12] = (_Float16)u3.x; b[13] = (_Float16)u3.y; b[14] = (_Float16)u3.z; b[15] = (_Float16)u3.w;
  return b;
}

// ---------------------------------------------------------------------------
// WMMA GEMM + fused epilogue:  Y[rows x cout] = ep( X[rows x cin] @ W[cout x cin]^T )
// Preconditions: cin % 32 == 0 (zero-padded), cout % 16 == 0, rows % 16 == 0,
//                MT==4 requires rows % 256 == 0.
// Each wave: MT 16x16 output tiles sharing one B fragment per K-step.
// mode 0: BN+ReLU ; 1: bias+ReLU ; 2: bias
// ---------------------------------------------------------------------------
template <int MT>
__global__ __launch_bounds__(128)
void gemm_ep_kernel(const float* __restrict__ X, const float* __restrict__ W,
                    const float* __restrict__ bias, const float* __restrict__ gam,
                    const float* __restrict__ beta, const float* __restrict__ mean,
                    const float* __restrict__ var,
                    float* __restrict__ Y, int rows, int cin, int cout, int mode)
{
  const int wave = threadIdx.x >> 5;
  const int lane = threadIdx.x & 31;
  const int m0 = (blockIdx.x * 4 + wave) * (16 * MT);
  if (m0 >= rows) return;                 // wave-uniform
  const int n0 = blockIdx.y * 16;
  const int gq = lane >> 4;               // half-wave group
  const int lr = lane & 15;
  const float* xbase = X + (size_t)(m0 + lr) * cin;   // A: M = lane%16
  const float* wrow  = W + (size_t)(n0 + lr) * cin;   // B: N = lane%16

  v8f acc[MT] = {};
  for (int k0 = 0; k0 < cin; k0 += 32) {
    const v16h bf = load_b_frag(wrow + k0 + 16 * gq);       // K = k0+16g+e
#pragma unroll
    for (int t = 0; t < MT; ++t) {
      const float* r = xbase + (size_t)(16 * t) * cin + k0;
      const v16h a = load_a_frag(r + 16 * gq, r + 16 * gq + 8); // K = k0+16g+e
      acc[t] = __builtin_amdgcn_wmma_f32_16x16x32_f16(false, a, false, bf,
                                                      (short)0, acc[t], false, false);
    }
  }

  const int col = n0 + lr;                // C/D: N = lane%16
  float scale = 1.f, shift = 0.f;
  if (mode == 0) {
    float sc = gam[col] * rsqrtf(var[col] + 1e-5f);
    scale = sc;
    shift = (bias[col] - mean[col]) * sc + beta[col];
  } else {
    shift = bias[col];
  }
#pragma unroll
  for (int t = 0; t < MT; ++t) {
#pragma unroll
    for (int r = 0; r < 8; ++r) {         // C/D: M = r + 8g
      int row = m0 + 16 * t + r + 8 * gq;
      float y = acc[t][r] * scale + shift;
      if (mode != 2) y = fmaxf(y, 0.f);
      Y[(size_t)row * cout + col] = y;
    }
  }
}

// --------------------------- pipeline kernels ------------------------------
__global__ void pad2d_kernel(const float* __restrict__ src, float* __restrict__ dst,
                             int srows, int scols, int drows, int dcols)
{
  int id = blockIdx.x * blockDim.x + threadIdx.x;
  if (id >= drows * dcols) return;
  int r = id / dcols, c = id % dcols;
  dst[id] = (r < srows && c < scols) ? src[(size_t)r * scols + c] : 0.f;
}

__global__ void transpose_xyz_kernel(const float* __restrict__ xyz,
                                     float* __restrict__ pts, int total, int n)
{
  int id = blockIdx.x * blockDim.x + threadIdx.x;
  if (id >= total) return;
  int b = id / n, i = id % n;
#pragma unroll
  for (int c = 0; c < 3; ++c)
    pts[(size_t)id * 3 + c] = xyz[((size_t)b * 3 + c) * n + i];
}

// Farthest point sampling, one block per batch, dist array in LDS (n<=4096).
__global__ __launch_bounds__(256)
void fps_kernel(const float* __restrict__ pts, int n, int npoint,
                int* __restrict__ outIdx)
{
  __shared__ float s_dist[4096];
  __shared__ float s_red[256];
  __shared__ int   s_redi[256];
  const float* p = pts + (size_t)blockIdx.x * n * 3;
  int* oi = outIdx + (size_t)blockIdx.x * npoint;
  const int t = threadIdx.x;
  for (int i = t; i < n; i += 256) s_dist[i] = 1e10f;
  __syncthreads();
  int far = 0;
  for (int s = 0; s < npoint; ++s) {
    if (t == 0) oi[s] = far;
    float cx = p[far * 3], cy = p[far * 3 + 1], cz = p[far * 3 + 2];
    float bm = -1.f; int bi = 0;
    for (int i = t; i < n; i += 256) {
      float dx = p[i * 3] - cx, dy = p[i * 3 + 1] - cy, dz = p[i * 3 + 2] - cz;
      float d = dx * dx + dy * dy + dz * dz;
      float dd = fminf(s_dist[i], d);
      s_dist[i] = dd;
      if (dd > bm) { bm = dd; bi = i; }
    }
    s_red[t] = bm; s_redi[t] = bi;
    __syncthreads();
    for (int off = 128; off > 0; off >>= 1) {
      if (t < off) {
        if (s_red[t + off] > s_red[t] ||
            (s_red[t + off] == s_red[t] && s_redi[t + off] < s_redi[t])) {
          s_red[t] = s_red[t + off]; s_redi[t] = s_redi[t + off];
        }
      }
      __syncthreads();
    }
    far = s_redi[0];
    __syncthreads();
  }
}

__global__ void gather_xyz_kernel(const float* __restrict__ src,
                                  const int* __restrict__ idx,
                                  float* __restrict__ dst,
                                  int total, int S, int Nsrc)
{
  int id = blockIdx.x * blockDim.x + threadIdx.x;
  if (id >= total) return;
  int b = id / S;
  int j = idx[id];
#pragma unroll
  for (int c = 0; c < 3; ++c)
    dst[(size_t)id * 3 + c] = src[((size_t)b * Nsrc + j) * 3 + c];
}

__global__ void ball_query_kernel(const float* __restrict__ xyz,
                                  const float* __restrict__ new_xyz,
                                  int* __restrict__ gidx,
                                  int total, int S, int N, int ns, float r2)
{
  int id = blockIdx.x * blockDim.x + threadIdx.x;
  if (id >= total) return;
  int b = id / S;
  const float* p = xyz + (size_t)b * N * 3;
  const float* q = new_xyz + (size_t)id * 3;
  float qx = q[0], qy = q[1], qz = q[2];
  int* o = gidx + (size_t)id * ns;
  int cnt = 0, first = 0;
  for (int n = 0; n < N; ++n) {
    float dx = p[n * 3] - qx, dy = p[n * 3 + 1] - qy, dz = p[n * 3 + 2] - qz;
    float d = dx * dx + dy * dy + dz * dz;
    if (d <= r2) {
      if (cnt == 0) first = n;
      o[cnt++] = n;
      if (cnt == ns) break;
    }
  }
  for (int k = cnt; k < ns; ++k) o[k] = first;
}

// rows: [ xyz[g]-new_xyz[s] (3) | feat[g] (C) | zeros.. (to ldo) ]  (per batch)
__global__ void group_concat_kernel(const float* __restrict__ xyz,
                                    const float* __restrict__ feat,
                                    const float* __restrict__ new_xyz,
                                    const int* __restrict__ gidx,
                                    float* __restrict__ out,
                                    int S, int ns, int C, int ldo)
{
  int row = blockIdx.x * blockDim.x + threadIdx.x;
  if (row >= S * ns) return;
  int s = row / ns;
  int g = gidx[row];
  const float* pg = xyz + (size_t)g * 3;
  const float* nx = new_xyz + (size_t)s * 3;
  float* o = out + (size_t)row * ldo;
  o[0] = pg[0] - nx[0]; o[1] = pg[1] - nx[1]; o[2] = pg[2] - nx[2];
  const float* f = feat + (size_t)g * C;
  for (int c = 0; c < C; ++c) o[3 + c] = f[c];
  for (int c = 3 + C; c < ldo; ++c) o[c] = 0.f;
}

__global__ void maxpool_kernel(const float* __restrict__ in,
                               float* __restrict__ out, int S, int ns, int C)
{
  int id = blockIdx.x * blockDim.x + threadIdx.x;
  if (id >= S * C) return;
  int s = id / C, c = id % C;
  const float* p = in + ((size_t)s * ns) * C + c;
  float m = p[0];
  for (int k = 1; k < ns; ++k) m = fmaxf(m, p[(size_t)k * C]);
  out[(size_t)s * C + c] = m;
}

// SA3 input rows: [l2_xyz(3) | l2_pts(256) | zeros(to 288)]
__global__ void sa3_concat_kernel(const float* __restrict__ l2xyz,
                                  const float* __restrict__ l2pts,
                                  float* __restrict__ out, int total)
{
  int r = blockIdx.x * blockDim.x + threadIdx.x;
  if (r >= total) return;
  float* o = out + (size_t)r * 288;
#pragma unroll
  for (int c = 0; c < 3; ++c) o[c] = l2xyz[(size_t)r * 3 + c];
  const float* f = l2pts + (size_t)r * 256;
  for (int c = 0; c < 256; ++c) o[3 + c] = f[c];
  for (int c = 259; c < 288; ++c) o[c] = 0.f;
}

// FP3 input rows: [l2_pts(256) | broadcast l3_pts(1024)]  (1280 % 32 == 0)
__global__ void fp3_concat_kernel(const float* __restrict__ l2pts,
                                  const float* __restrict__ l3pts,
                                  float* __restrict__ out, int total)
{
  int r = blockIdx.x * blockDim.x + threadIdx.x;
  if (r >= total) return;
  int b = r / 128;
  float* o = out + (size_t)r * 1280;
  const float* f1 = l2pts + (size_t)r * 256;
  for (int c = 0; c < 256; ++c) o[c] = f1[c];
  const float* f2 = l3pts + (size_t)b * 1024;
  for (int c = 0; c < 1024; ++c) o[256 + c] = f2[c];
}

// 3-NN with inverse-distance weights (matches top_k(-d,3) then 1/(d+1e-8))
__global__ void knn3_kernel(const float* __restrict__ xyz1,
                            const float* __restrict__ xyz2,
                            int* __restrict__ kidx, float* __restrict__ kw,
                            int total, int N1, int N2)
{
  int id = blockIdx.x * blockDim.x + threadIdx.x;
  if (id >= total) return;
  int b = id / N1;
  const float* q = xyz1 + (size_t)id * 3;
  float qx = q[0], qy = q[1], qz = q[2];
  const float* p = xyz2 + (size_t)b * N2 * 3;
  float d0 = 1e30f, d1 = 1e30f, d2 = 1e30f;
  int i0 = 0, i1 = 0, i2 = 0;
  for (int n = 0; n < N2; ++n) {
    float dx = p[n * 3] - qx, dy = p[n * 3 + 1] - qy, dz = p[n * 3 + 2] - qz;
    float d = dx * dx + dy * dy + dz * dz;
    if (d < d0)      { d2 = d1; i2 = i1; d1 = d0; i1 = i0; d0 = d; i0 = n; }
    else if (d < d1) { d2 = d1; i2 = i1; d1 = d;  i1 = n; }
    else if (d < d2) { d2 = d;  i2 = n; }
  }
  float w0 = 1.f / (d0 + 1e-8f), w1 = 1.f / (d1 + 1e-8f), w2 = 1.f / (d2 + 1e-8f);
  float inv = 1.f / (w0 + w1 + w2);
  kidx[(size_t)id * 3 + 0] = i0; kidx[(size_t)id * 3 + 1] = i1; kidx[(size_t)id * 3 + 2] = i2;
  kw[(size_t)id * 3 + 0] = w0 * inv; kw[(size_t)id * 3 + 1] = w1 * inv; kw[(size_t)id * 3 + 2] = w2 * inv;
}

// FP concat: [ feat1 (C1, or dup'd xyz if dup) | 3NN-interp feat2 (C2) | zeros..ldo ]
__global__ void fp_concat_kernel(const float* __restrict__ feat1, int C1, int dup,
                                 const float* __restrict__ feat2, int C2, int N2,
                                 const int* __restrict__ kidx,
                                 const float* __restrict__ kw,
                                 float* __restrict__ out, int total, int N1, int ldo)
{
  int r = blockIdx.x * blockDim.x + threadIdx.x;
  if (r >= total) return;
  int b = r / N1;
  float* o = out + (size_t)r * ldo;
  if (dup) {
    for (int c = 0; c < C1; ++c) o[c] = feat1[(size_t)r * 3 + (c % 3)];
  } else {
    const float* f1 = feat1 + (size_t)r * C1;
    for (int c = 0; c < C1; ++c) o[c] = f1[c];
  }
  const int* ki = kidx + (size_t)r * 3;
  const float* w = kw + (size_t)r * 3;
  const float* f2 = feat2 + (size_t)b * N2 * C2;
  const float* p0 = f2 + (size_t)ki[0] * C2;
  const float* p1 = f2 + (size_t)ki[1] * C2;
  const float* p2 = f2 + (size_t)ki[2] * C2;
  float w0 = w[0], w1 = w[1], w2 = w[2];
  for (int c = 0; c < C2; ++c)
    o[C1 + c] = w0 * p0[c] + w1 * p1[c] + w2 * p2[c];
  for (int c = C1 + C2; c < ldo; ++c) o[c] = 0.f;
}

__global__ void logsoftmax_kernel(const float* __restrict__ x,
                                  float* __restrict__ y, int rows, int C)
{
  int r = blockIdx.x * blockDim.x + threadIdx.x;
  if (r >= rows) return;
  const float* xr = x + (size_t)r * C;
  float mx = -1e30f;
  for (int c = 0; c < C; ++c) mx = fmaxf(mx, xr[c]);
  float s = 0.f;
  for (int c = 0; c < C; ++c) s += expf(xr[c] - mx);
  float ls = logf(s);
  for (int c = 0; c < C; ++c) y[(size_t)r * C + c] = xr[c] - mx - ls;
}

// softmax over n (axis=1) fused with einsum('bnc,bnj->bjc'): block per (b,j)
__global__ __launch_bounds__(256)
void joint_kernel(const float* __restrict__ logits, const float* __restrict__ pts,
                  float* __restrict__ out, int n, int ldJ, int J)
{
  __shared__ float red[256];
  const int b = blockIdx.x, j = blockIdx.y, t = threadIdx.x;
  const float* lg = logits + (size_t)b * n * ldJ + j;
  float mx = -1e30f;
  for (int i = t; i < n; i += 256) mx = fmaxf(mx, lg[(size_t)i * ldJ]);
  red[t] = mx; __syncthreads();
  for (int o = 128; o > 0; o >>= 1) { if (t < o) red[t] = fmaxf(red[t], red[t + o]); __syncthreads(); }
  float M = red[0]; __syncthreads();
  const float* pb = pts + (size_t)b * n * 3;
  float se = 0.f, sx = 0.f, sy = 0.f, sz = 0.f;
  for (int i = t; i < n; i += 256) {
    float e = expf(lg[(size_t)i * ldJ] - M);
    se += e;
    sx += e * pb[(size_t)i * 3];
    sy += e * pb[(size_t)i * 3 + 1];
    sz += e * pb[(size_t)i * 3 + 2];
  }
  float vals[4] = { se, sx, sy, sz };
  float res[4];
#pragma unroll
  for (int k = 0; k < 4; ++k) {
    red[t] = vals[k]; __syncthreads();
    for (int o = 128; o > 0; o >>= 1) { if (t < o) red[t] += red[t + o]; __syncthreads(); }
    res[k] = red[0]; __syncthreads();
  }
  if (t == 0) {
    float inv = 1.f / res[0];
    float* o = out + ((size_t)b * J + j) * 3;
    o[0] = res[1] * inv; o[1] = res[2] * inv; o[2] = res[3] * inv;
  }
}

// ------------------------------- host side ---------------------------------
struct LayerP { const float *W, *b, *g, *be, *m, *v; };
static inline int cdiv(int a, int b) { return (a + b - 1) / b; }

extern "C" void kernel_launch(void* const* d_in, const int* in_sizes, int n_in,
                              void* d_out, int out_size, void* d_ws, size_t ws_size,
                              hipStream_t stream)
{
  (void)in_sizes; (void)n_in; (void)out_size; (void)ws_size;
  auto F  = [&](int i) { return (const float*)d_in[i]; };
  auto L6 = [&](int i) { LayerP l = { F(i), F(i+1), F(i+2), F(i+3), F(i+4), F(i+5) }; return l; };

  const float* xyz = F(0);
  LayerP sa1[3] = { L6(1),  L6(7),  L6(13) };
  LayerP sa2[3] = { L6(19), L6(25), L6(31) };
  LayerP sa3[3] = { L6(37), L6(43), L6(49) };
  LayerP fp3[2] = { L6(55), L6(61) };
  LayerP fp2[2] = { L6(67), L6(73) };
  LayerP fp1[3] = { L6(79), L6(85), L6(91) };
  LayerP seg1   = L6(97);
  LayerP seg2   = { F(103), F(104), nullptr, nullptr, nullptr, nullptr };
  LayerP cls0   = { F(105), F(106), nullptr, nullptr, nullptr, nullptr };
  LayerP cls1   = { F(107), F(108), nullptr, nullptr, nullptr, nullptr };
  LayerP cls2   = { F(109), F(110), nullptr, nullptr, nullptr, nullptr };

  // ---- workspace bump allocator (all chunks 256B aligned) ----
  size_t off = 0;
  auto alloc = [&](size_t bytes) -> void* {
    void* p = (char*)d_ws + off;
    off += (bytes + 255) & ~(size_t)255;
    return p;
  };
  float* pts   = (float*)alloc((size_t)BB * NPTS * 3 * 4);
  float* pingA = (float*)alloc((size_t)65536 * 160 * 4);   // largest padded GEMM input
  float* pingB = (float*)alloc((size_t)65536 * 128 * 4);
  int*   fidx1 = (int*)  alloc((size_t)BB * 512 * 4);
  float* l1xyz = (float*)alloc((size_t)BB * 512 * 3 * 4);
  int*   gidx1 = (int*)  alloc((size_t)BB * 512 * 32 * 4);
  float* l1pts = (float*)alloc((size_t)BB * 512 * 128 * 4);
  int*   fidx2 = (int*)  alloc((size_t)BB * 128 * 4);
  float* l2xyz = (float*)alloc((size_t)BB * 128 * 3 * 4);
  int*   gidx2 = (int*)  alloc((size_t)BB * 128 * 64 * 4);
  float* l2pts = (float*)alloc((size_t)BB * 128 * 256 * 4);
  float* l3pts = (float*)alloc((size_t)BB * 1024 * 4);
  float* clsh1 = (float*)alloc((size_t)BB * 512 * 4);
  float* clsh2 = (float*)alloc((size_t)BB * 256 * 4);
  float* clsl  = (float*)alloc((size_t)BB * 16 * 4);
  float* l2f   = (float*)alloc((size_t)BB * 128 * 256 * 4);
  float* l1f   = (float*)alloc((size_t)BB * 512 * 128 * 4);
  int*   kidx  = (int*)  alloc((size_t)BB * NPTS * 3 * 4);
  float* kw    = (float*)alloc((size_t)BB * NPTS * 3 * 4);
  float* logits= (float*)alloc((size_t)BB * NPTS * 32 * 4);   // ld = 32 (24 used)
  float* sa1w0 = (float*)alloc((size_t)64  * 32  * 4);
  float* sa2w0 = (float*)alloc((size_t)128 * 160 * 4);
  float* sa3w0 = (float*)alloc((size_t)256 * 288 * 4);
  float* fp1w0 = (float*)alloc((size_t)128 * 160 * 4);
  float* seg2wp= (float*)alloc((size_t)32  * 128 * 4);
  float* seg2bp= (float*)alloc((size_t)32 * 4);

  // ---- zero-pad ragged weight matrices once (cin -> mult of 32, cout -> 16) ----
  auto pad = [&](const float* src, float* dst, int sr, int sc, int dr, int dc) {
    pad2d_kernel<<<cdiv(dr * dc, 256), 256, 0, stream>>>(src, dst, sr, sc, dr, dc);
  };
  pad(sa1[0].W, sa1w0, 64, 6,   64, 32);
  pad(sa2[0].W, sa2w0, 128, 131, 128, 160);
  pad(sa3[0].W, sa3w0, 256, 259, 256, 288);
  pad(fp1[0].W, fp1w0, 128, 134, 128, 160);
  pad(seg2.W,   seg2wp, 24, 128, 32, 128);
  pad(seg2.b,   seg2bp, 24, 1,   32, 1);
  LayerP sa1l0 = sa1[0]; sa1l0.W = sa1w0;
  LayerP sa2l0 = sa2[0]; sa2l0.W = sa2w0;
  LayerP sa3l0 = sa3[0]; sa3l0.W = sa3w0;
  LayerP fp1l0 = fp1[0]; fp1l0.W = fp1w0;
  LayerP seg2p = { seg2wp, seg2bp, nullptr, nullptr, nullptr, nullptr };

  auto gemm = [&](const float* X, const LayerP& l, float* Y,
                  int rows, int cin, int cout, int mode) {
    if (rows % 256 == 0) {
      dim3 grid(rows / 256, cout / 16);
      gemm_ep_kernel<4><<<grid, 128, 0, stream>>>(X, l.W, l.b, l.g, l.be, l.m, l.v,
                                                  Y, rows, cin, cout, mode);
    } else {
      dim3 grid(cdiv(rows, 64), cout / 16);
      gemm_ep_kernel<1><<<grid, 128, 0, stream>>>(X, l.W, l.b, l.g, l.be, l.m, l.v,
                                                  Y, rows, cin, cout, mode);
    }
  };

  // ---- l0: transpose (B,3,N) -> (B,N,3) ----
  transpose_xyz_kernel<<<cdiv(BB * NPTS, 256), 256, 0, stream>>>(xyz, pts, BB * NPTS, NPTS);

  // ---- SA1: fps 512, ball r=0.2 ns=32, MLP 6->64->64->128, max over 32 ----
  fps_kernel<<<BB, 256, 0, stream>>>(pts, NPTS, 512, fidx1);
  gather_xyz_kernel<<<cdiv(BB * 512, 256), 256, 0, stream>>>(pts, fidx1, l1xyz, BB * 512, 512, NPTS);
  ball_query_kernel<<<cdiv(BB * 512, 256), 256, 0, stream>>>(pts, l1xyz, gidx1,
                                                             BB * 512, 512, NPTS, 32, 0.2f * 0.2f);
  for (int b = 0; b < BB; ++b) {
    const float* pb = pts + (size_t)b * NPTS * 3;
    group_concat_kernel<<<cdiv(512 * 32, 256), 256, 0, stream>>>(
        pb, pb, l1xyz + (size_t)b * 512 * 3, gidx1 + (size_t)b * 512 * 32,
        pingA, 512, 32, 3, 32);
    gemm(pingA, sa1l0,  pingB, 512 * 32, 32, 64, 0);
    gemm(pingB, sa1[1], pingA, 512 * 32, 64, 64, 0);
    gemm(pingA, sa1[2], pingB, 512 * 32, 64, 128, 0);
    maxpool_kernel<<<cdiv(512 * 128, 256), 256, 0, stream>>>(
        pingB, l1pts + (size_t)b * 512 * 128, 512, 32, 128);
  }

  // ---- SA2: fps 128 of 512, ball r=0.4 ns=64, MLP 131->128->128->256 ----
  fps_kernel<<<BB, 256, 0, stream>>>(l1xyz, 512, 128, fidx2);
  gather_xyz_kernel<<<cdiv(BB * 128, 256), 256, 0, stream>>>(l1xyz, fidx2, l2xyz, BB * 128, 128, 512);
  ball_query_kernel<<<cdiv(BB * 128, 256), 256, 0, stream>>>(l1xyz, l2xyz, gidx2,
                                                             BB * 128, 128, 512, 64, 0.4f * 0.4f);
  for (int b = 0; b < BB; ++b) {
    group_concat_kernel<<<cdiv(128 * 64, 256), 256, 0, stream>>>(
        l1xyz + (size_t)b * 512 * 3, l1pts + (size_t)b * 512 * 128,
        l2xyz + (size_t)b * 128 * 3, gidx2 + (size_t)b * 128 * 64,
        pingA, 128, 64, 128, 160);
    gemm(pingA, sa2l0,  pingB, 128 * 64, 160, 128, 0);
    gemm(pingB, sa2[1], pingA, 128 * 64, 128, 128, 0);
    gemm(pingA, sa2[2], pingB, 128 * 64, 128, 256, 0);
    maxpool_kernel<<<cdiv(128 * 256, 256), 256, 0, stream>>>(
        pingB, l2pts + (size_t)b * 128 * 256, 128, 64, 256);
  }

  // ---- SA3 (group_all): 259(->288) -> 256 -> 512 -> 1024, max over 128 ----
  sa3_concat_kernel<<<cdiv(BB * 128, 128), 128, 0, stream>>>(l2xyz, l2pts, pingA, BB * 128);
  gemm(pingA, sa3l0,  pingB, BB * 128, 288, 256, 0);
  gemm(pingB, sa3[1], pingA, BB * 128, 256, 512, 0);
  gemm(pingA, sa3[2], pingB, BB * 128, 512, 1024, 0);
  maxpool_kernel<<<cdiv(BB * 1024, 256), 256, 0, stream>>>(pingB, l3pts, BB, 128, 1024);

  // ---- classifier head: 1024->512->256->16, log_softmax ----
  gemm(l3pts, cls0, clsh1, BB, 1024, 512, 1);
  gemm(clsh1, cls1, clsh2, BB, 512, 256, 1);
  gemm(clsh2, cls2, clsl,  BB, 256, 16, 2);
  logsoftmax_kernel<<<1, 32, 0, stream>>>(clsl, (float*)d_out + BB * 24 * 3, BB, 16);

  // ---- FP3 (S==1 broadcast): [l2_pts(256)|l3_pts(1024)] -> 256 -> 256 ----
  fp3_concat_kernel<<<cdiv(BB * 128, 128), 128, 0, stream>>>(l2pts, l3pts, pingA, BB * 128);
  gemm(pingA, fp3[0], pingB, BB * 128, 1280, 256, 0);
  gemm(pingB, fp3[1], l2f,   BB * 128, 256, 256, 0);

  // ---- FP2: 3NN(l1_xyz <- l2_xyz), [l1_pts(128)|interp(256)] -> 256 -> 128 ----
  knn3_kernel<<<cdiv(BB * 512, 256), 256, 0, stream>>>(l1xyz, l2xyz, kidx, kw, BB * 512, 512, 128);
  fp_concat_kernel<<<cdiv(BB * 512, 128), 128, 0, stream>>>(
      l1pts, 128, 0, l2f, 256, 128, kidx, kw, pingA, BB * 512, 512, 384);
  gemm(pingA, fp2[0], pingB, BB * 512, 384, 256, 0);
  gemm(pingB, fp2[1], l1f,   BB * 512, 256, 128, 0);

  // ---- FP1: 3NN(pts <- l1_xyz), [pts,pts(6)|interp(128)](->160) ->128 ->128 ->128 ----
  knn3_kernel<<<cdiv(BB * NPTS, 256), 256, 0, stream>>>(pts, l1xyz, kidx, kw, BB * NPTS, NPTS, 512);
  fp_concat_kernel<<<cdiv(BB * NPTS, 128), 128, 0, stream>>>(
      pts, 6, 1, l1f, 128, 512, kidx, kw, pingA, BB * NPTS, NPTS, 160);
  gemm(pingA, fp1l0,  pingB, BB * NPTS, 160, 128, 0);
  gemm(pingB, fp1[1], pingA, BB * NPTS, 128, 128, 0);
  gemm(pingA, fp1[2], pingB, BB * NPTS, 128, 128, 0);   // l0_f in pingB

  // ---- seg head: BN layer 128->128, linear 128->24(padded 32) ----
  gemm(pingB, seg1,  pingA, BB * NPTS, 128, 128, 0);
  gemm(pingA, seg2p, logits, BB * NPTS, 128, 32, 2);

  // ---- softmax over N + einsum('bnc,bnj->bjc') -> joint_pos (B,24,3) ----
  joint_kernel<<<dim3(BB, 24), 256, 0, stream>>>(logits, pts, (float*)d_out, NPTS, 32, 24);
}
